// Binary_CNN1_58892591563200
// MI455X (gfx1250) — 8-bit matrix operand order repaired and checked by reading, compiled, not run on hardware
//
#include <hip/hip_runtime.h>
#include <hip/hip_bf16.h>
#include <stdint.h>

typedef __attribute__((ext_vector_type(8))) int   v8i_t;
typedef __attribute__((ext_vector_type(8))) float v8f_t;
typedef __attribute__((ext_vector_type(2))) float v2f_t;

#define BN_EPS 1e-5f

__global__ void fill_zero16_kernel(uint4* __restrict__ p, long long n16) {
  long long i = (long long)blockIdx.x * blockDim.x + threadIdx.x;
  if (i < n16) p[i] = make_uint4(0u, 0u, 0u, 0u);
}

__global__ void pack_w_conv_kernel(const float* __restrict__ w, signed char* __restrict__ dst,
                                   int Cout, int Cin, int Kp) {
  long long i = (long long)blockIdx.x * blockDim.x + threadIdx.x;
  long long total = (long long)Cout * Cin * 9;
  if (i >= total) return;
  int kw = (int)(i % 3); long long r = i / 3;
  int kh = (int)(r % 3); r /= 3;
  int c  = (int)(r % Cin);
  int co = (int)(r / Cin);
  dst[(long long)co * Kp + (kh * 3 + kw) * Cin + c] = (w[i] >= 0.f) ? 1 : -1;
}

__global__ void pack_sign_kernel(const float* __restrict__ w, signed char* __restrict__ dst,
                                 long long n) {
  long long i = (long long)blockIdx.x * blockDim.x + threadIdx.x;
  if (i < n) dst[i] = (w[i] >= 0.f) ? 1 : -1;
}

__global__ void conv1_pool_kernel(const float* __restrict__ x, const float* __restrict__ w1,
                                  const float* __restrict__ b1, float* __restrict__ out) {
  long long i = (long long)blockIdx.x * blockDim.x + threadIdx.x;
  const long long total = 1024LL * 14 * 14 * 48;
  if (i >= total) return;
  int co = (int)(i % 48); long long r = i / 48;
  int ow = (int)(r % 14); r /= 14;
  int oh = (int)(r % 14);
  int b  = (int)(r / 14);
  bool ws[9];
#pragma unroll
  for (int t = 0; t < 9; ++t) ws[t] = (w1[co * 9 + t] >= 0.f);
  const float* xb = x + (long long)b * 784;
  float best = -3.4e38f;
#pragma unroll
  for (int dy = 0; dy < 2; ++dy)
#pragma unroll
    for (int dx = 0; dx < 2; ++dx) {
      int ch = 2 * oh + dy, cw = 2 * ow + dx;
      int s = 0;
#pragma unroll
      for (int kh = 0; kh < 3; ++kh)
#pragma unroll
        for (int kw = 0; kw < 3; ++kw) {
          int ih = ch + kh - 1, iw = cw + kw - 1;
          if ((unsigned)ih < 28u && (unsigned)iw < 28u) {
            bool xs = (xb[ih * 28 + iw] >= 0.f);
            s += (xs == ws[kh * 3 + kw]) ? 1 : -1;
          }
        }
      best = fmaxf(best, (float)s + b1[co]);
    }
  out[i] = best;
}

__global__ void bn_stats_kernel(const float* __restrict__ X, float* __restrict__ st,
                                long long M, int C) {
  __shared__ float s1[256];
  __shared__ float s2[256];
  int c = blockIdx.x;
  float a = 0.f, b = 0.f;
  for (long long m = (long long)blockIdx.y * blockDim.x + threadIdx.x; m < M;
       m += (long long)gridDim.y * blockDim.x) {
    float v = X[m * C + c];
    a += v; b += v * v;
  }
  int t = threadIdx.x;
  s1[t] = a; s2[t] = b;
  __syncthreads();
  for (int s = 128; s > 0; s >>= 1) {
    if (t < s) { s1[t] += s1[t + s]; s2[t] += s2[t + s]; }
    __syncthreads();
  }
  if (t == 0) { atomicAdd(&st[c], s1[0]); atomicAdd(&st[2048 + c], s2[0]); }
}

__global__ void bn_htanh_bin_kernel(const float* __restrict__ X, const float* __restrict__ st,
                                    const float* __restrict__ g, const float* __restrict__ be,
                                    signed char* __restrict__ out, long long M, int C) {
  long long i = (long long)blockIdx.x * blockDim.x + threadIdx.x;
  if (i >= M * (long long)C) return;
  int c = (int)(i % C);
  float mean = st[c] / (float)M;
  float var  = st[2048 + c] / (float)M - mean * mean;
  float v = (X[i] - mean) * rsqrtf(var + BN_EPS) * g[c] + be[c];
  out[i] = (v >= 0.f) ? 1 : -1;
}

__global__ void bn_htanh_bin_reorder_kernel(const float* __restrict__ X, const float* __restrict__ st,
                                            const float* __restrict__ g, const float* __restrict__ be,
                                            signed char* __restrict__ out) {
  long long i = (long long)blockIdx.x * blockDim.x + threadIdx.x;
  const long long total = 1024LL * 49 * 256;
  if (i >= total) return;
  int c = (int)(i % 256); long long r = i / 256;
  int p = (int)(r % 49);
  int b = (int)(r / 49);
  const float Mf = 50176.f;
  float mean = st[c] / Mf;
  float var  = st[2048 + c] / Mf - mean * mean;
  float v = (X[i] - mean) * rsqrtf(var + BN_EPS) * g[c] + be[c];
  out[(long long)b * 12544 + c * 49 + p] = (v >= 0.f) ? 1 : -1;
}

__global__ void bn_htanh_f_kernel(float* __restrict__ X, const float* __restrict__ st,
                                  const float* __restrict__ g, const float* __restrict__ be,
                                  long long M, int C) {
  long long i = (long long)blockIdx.x * blockDim.x + threadIdx.x;
  if (i >= M * (long long)C) return;
  int c = (int)(i % C);
  float mean = st[c] / (float)M;
  float var  = st[2048 + c] / (float)M - mean * mean;
  float v = (X[i] - mean) * rsqrtf(var + BN_EPS) * g[c] + be[c];
  X[i] = fminf(1.f, fmaxf(-1.f, v));
}

__global__ void im2col3x3_kernel(const signed char* __restrict__ src, signed char* __restrict__ dst,
                                 int Bn, int H, int W, int C, int Kp) {
  long long t = (long long)blockIdx.x * blockDim.x + threadIdx.x;
  long long total = (long long)Bn * H * W * 9;
  if (t >= total) return;
  int tap = (int)(t % 9); long long pix = t / 9;
  int ow = (int)(pix % W); long long r = pix / W;
  int oh = (int)(r % H);
  int b  = (int)(r / H);
  int kh = tap / 3, kw = tap % 3;
  int ih = oh + kh - 1, iw = ow + kw - 1;
  signed char* out = dst + pix * (long long)Kp + (long long)tap * C;
  if ((unsigned)ih < (unsigned)H && (unsigned)iw < (unsigned)W) {
    const signed char* in = src + (((long long)b * H + ih) * W + iw) * C;
    for (int c = 0; c < C; c += 16) *(int4*)(out + c) = *(const int4*)(in + c);
  } else {
    const int4 z = make_int4(0, 0, 0, 0);
    for (int c = 0; c < C; c += 16) *(int4*)(out + c) = z;
  }
}

__global__ void maxpool2_nhwc_kernel(const float* __restrict__ in, float* __restrict__ out,
                                     int Bn, int H, int W, int C) {
  long long i = (long long)blockIdx.x * blockDim.x + threadIdx.x;
  long long total = (long long)Bn * (H / 2) * (W / 2) * C;
  if (i >= total) return;
  int c = (int)(i % C); long long r = i / C;
  int ow = (int)(r % (W / 2)); r /= (W / 2);
  int oh = (int)(r % (H / 2));
  int b  = (int)(r / (H / 2));
  const float* p = in + (((long long)b * H + 2 * oh) * W + 2 * ow) * C + c;
  long long rowstride = (long long)W * C;
  out[i] = fmaxf(fmaxf(p[0], p[C]), fmaxf(p[rowstride], p[rowstride + C]));
}

__global__ __launch_bounds__(256) void bgemm_iu8_kernel(
    const signed char* __restrict__ A, const signed char* __restrict__ Bp,
    const float* __restrict__ bias, float* __restrict__ C,
    int M, int N, int K) {
  __shared__ __align__(16) signed char sB[2][128 * 64];

  const int tid   = threadIdx.x;
  const int lane  = tid & 31;
  const int wave  = tid >> 5;
  const int wrow  = wave & 3;
  const int wcol  = wave >> 2;
  const int row16 = lane & 15;
  const int half  = lane >> 4;

  const int mtile = blockIdx.y * 4 + wrow;
  const int nbase = blockIdx.x * 128;
  const long long m = (long long)mtile * 16 + row16;
  const signed char* arow = A + m * (long long)K;

  const int nl0 = tid >> 2,         ko0 = (tid & 3) * 16;
  const int nl1 = (tid + 256) >> 2, ko1 = ((tid + 256) & 3) * 16;

  v8i_t acc0 = {}, acc1 = {}, acc2 = {}, acc3 = {};
  const int nchunks = K >> 6;

  {
    unsigned long long g0 = (unsigned long long)(Bp + (long long)(nbase + nl0) * K + ko0);
    unsigned long long g1 = (unsigned long long)(Bp + (long long)(nbase + nl1) * K + ko1);
    unsigned l0 = (unsigned)(uintptr_t)&sB[0][nl0 * 64 + ko0];
    unsigned l1 = (unsigned)(uintptr_t)&sB[0][nl1 * 64 + ko1];
    asm volatile("global_load_async_to_lds_b128 %0, %1, off" ::"v"(l0), "v"(g0) : "memory");
    asm volatile("global_load_async_to_lds_b128 %0, %1, off" ::"v"(l1), "v"(g1) : "memory");
  }

  for (int i = 0; i < nchunks; ++i) {
    const int k0 = i << 6;
    const signed char* ap = arow + k0 + half * 16;
    int2 s0 = *(const int2*)(ap);
    int2 s1 = *(const int2*)(ap + 8);
    int2 s2 = *(const int2*)(ap + 32);
    int2 s3 = *(const int2*)(ap + 40);

    if (i + 1 < nchunks) {
      const int kn = k0 + 64, nb = (i + 1) & 1;
      unsigned long long g0 = (unsigned long long)(Bp + (long long)(nbase + nl0) * K + kn + ko0);
      unsigned long long g1 = (unsigned long long)(Bp + (long long)(nbase + nl1) * K + kn + ko1);
      unsigned l0 = (unsigned)(uintptr_t)&sB[nb][nl0 * 64 + ko0];
      unsigned l1 = (unsigned)(uintptr_t)&sB[nb][nl1 * 64 + ko1];
      asm volatile("global_load_async_to_lds_b128 %0, %1, off" ::"v"(l0), "v"(g0) : "memory");
      asm volatile("global_load_async_to_lds_b128 %0, %1, off" ::"v"(l1), "v"(g1) : "memory");
      asm volatile("s_wait_asynccnt 2" ::: "memory");
    } else {
      asm volatile("s_wait_asynccnt 0" ::: "memory");
    }
    __syncthreads();

    v8i_t a;
    a[0] = s0.x; a[1] = s0.y; a[2] = s1.x; a[3] = s1.y;
    a[4] = s2.x; a[5] = s2.y; a[6] = s3.x; a[7] = s3.y;

    const signed char* bcol = &sB[i & 1][(wcol * 64 + row16) * 64 + half * 16];
#pragma unroll
    for (int t = 0; t < 4; ++t) {
      const signed char* bp = bcol + t * (16 * 64);
      int4 t0 = *(const int4*)(bp);
      int4 t1 = *(const int4*)(bp + 32);
      v8i_t b;
      b[0] = t0.x; b[1] = t0.y; b[2] = t0.z; b[3] = t0.w;
      b[4] = t1.x; b[5] = t1.y; b[6] = t1.z; b[7] = t1.w;
      if (t == 0) acc0 = __builtin_amdgcn_wmma_i32_16x16x64_iu8(true, a, true, b, acc0, false, false);
      if (t == 1) acc1 = __builtin_amdgcn_wmma_i32_16x16x64_iu8(true, a, true, b, acc1, false, false);
      if (t == 2) acc2 = __builtin_amdgcn_wmma_i32_16x16x64_iu8(true, a, true, b, acc2, false, false);
      if (t == 3) acc3 = __builtin_amdgcn_wmma_i32_16x16x64_iu8(true, a, true, b, acc3, false, false);
    }
    __syncthreads();
  }

#pragma unroll
  for (int t = 0; t < 4; ++t) {
    const v8i_t* accp = (t == 0) ? &acc0 : (t == 1) ? &acc1 : (t == 2) ? &acc2 : &acc3;
    const int n = nbase + wcol * 64 + t * 16 + row16;
    const float bv = bias[n];
#pragma unroll
    for (int rr = 0; rr < 8; ++rr) {
      long long mr = (long long)mtile * 16 + half * 8 + rr;
      C[mr * N + n] = (float)((*accp)[rr]) + bv;
    }
  }
}

__global__ __launch_bounds__(256) void fc2_wmma_kernel(const float* __restrict__ A,
                                                       const float* __restrict__ Wt,
                                                       const float* __restrict__ bias,
                                                       float* __restrict__ out) {
  const int lane  = threadIdx.x & 31;
  const int wave  = threadIdx.x >> 5;
  const int mtile = blockIdx.x * 8 + wave;
  const int row16 = lane & 15;
  const int half  = lane >> 4;
  const int m  = mtile * 16 + row16;
  const int n  = row16;
  const int nc = (n < 10) ? n : 9;
  const float* arow = A + (long long)m * 2048;
  const float* brow = Wt + (long long)nc * 2048;
  v8f_t acc = {};
  for (int k = 0; k < 2048; k += 4) {
    v2f_t a = *(const v2f_t*)(arow + k + half * 2);
    v2f_t b = *(const v2f_t*)(brow + k + half * 2);
    acc = __builtin_amdgcn_wmma_f32_16x16x4_f32(false, a, false, b, (short)0, acc, false, false);
  }
  if (n < 10) {
    const float bv = bias[n];
#pragma unroll
    for (int rr = 0; rr < 8; ++rr) {
      int mr = mtile * 16 + half * 8 + rr;
      out[(long long)mr * 10 + n] = acc[rr] + bv;
    }
  }
}

__global__ void logsoftmax10_kernel(float* __restrict__ io, int M) {
  int m = blockIdx.x * blockDim.x + threadIdx.x;
  if (m >= M) return;
  float v[10]; float mx = -3.4e38f;
#pragma unroll
  for (int j = 0; j < 10; ++j) { v[j] = io[m * 10 + j]; mx = fmaxf(mx, v[j]); }
  float s = 0.f;
#pragma unroll
  for (int j = 0; j < 10; ++j) s += __expf(v[j] - mx);
  float ls = __logf(s);
#pragma unroll
  for (int j = 0; j < 10; ++j) io[m * 10 + j] = v[j] - mx - ls;
}

extern "C" void kernel_launch(void* const* d_in, const int* in_sizes, int n_in,
                              void* d_out, int out_size, void* d_ws, size_t ws_size,
                              hipStream_t stream) {
  (void)in_sizes; (void)n_in; (void)out_size; (void)ws_size;
  const float* x   = (const float*)d_in[0];
  const float* w1  = (const float*)d_in[1];
  const float* b1  = (const float*)d_in[2];
  const float* g1  = (const float*)d_in[3];
  const float* be1 = (const float*)d_in[4];
  const float* w2  = (const float*)d_in[5];
  const float* b2  = (const float*)d_in[6];
  const float* g2  = (const float*)d_in[7];
  const float* be2 = (const float*)d_in[8];
  const float* w3  = (const float*)d_in[9];
  const float* b3  = (const float*)d_in[10];
  const float* g3  = (const float*)d_in[11];
  const float* be3 = (const float*)d_in[12];
  const float* wf1 = (const float*)d_in[13];
  const float* bf1 = (const float*)d_in[14];
  const float* g4  = (const float*)d_in[15];
  const float* be4 = (const float*)d_in[16];
  const float* wf2 = (const float*)d_in[17];
  const float* bf2 = (const float*)d_in[18];
  float* outp = (float*)d_out;

  const long long M2 = 200704;
  const int K2P = 448;
  const int K3P = 1152;
  const long long K4 = 12544;

  char* ws = (char*)d_ws;
  size_t off = 0;
  const size_t OFF_R0 = off; off += 205520896;
  const size_t OFF_R1 = off; off += 231211008;
  const size_t OFF_B2 = off; off += 57344;
  const size_t OFF_B3 = off; off += 294912;
  const size_t OFF_B4 = off; off += (size_t)2048 * 12544;
  const size_t OFF_HB2 = off; off += (size_t)M2 * 48;
  const size_t OFF_HB3 = off; off += (size_t)M2 * 128;
  const size_t OFF_ST = off; off += 16384;

  signed char* b2p = (signed char*)(ws + OFF_B2);
  signed char* b3p = (signed char*)(ws + OFF_B3);
  signed char* b4p = (signed char*)(ws + OFF_B4);
  signed char* hb2 = (signed char*)(ws + OFF_HB2);
  signed char* hb3 = (signed char*)(ws + OFF_HB3);
  float*       st  = (float*)(ws + OFF_ST);

  auto nb = [](long long n, int b) { return (unsigned)((n + b - 1) / b); };
  const int T = 256;

  pack_w_conv_kernel<<<nb(128LL * 48 * 9, T), T, 0, stream>>>(w2, b2p, 128, 48, K2P);
  pack_w_conv_kernel<<<nb(256LL * 128 * 9, T), T, 0, stream>>>(w3, b3p, 256, 128, K3P);
  pack_sign_kernel<<<nb(2048LL * 12544, T), T, 0, stream>>>(wf1, b4p, 2048LL * 12544);

  float* h1p = (float*)(ws + OFF_R0);
  conv1_pool_kernel<<<nb(M2 * 48, T), T, 0, stream>>>(x, w1, b1, h1p);
  fill_zero16_kernel<<<4, T, 0, stream>>>((uint4*)st, 1024);
  bn_stats_kernel<<<dim3(48, 64), T, 0, stream>>>(h1p, st, M2, 48);
  bn_htanh_bin_kernel<<<nb(M2 * 48, T), T, 0, stream>>>(h1p, st, g1, be1, hb2, M2, 48);

  signed char* a2 = (signed char*)(ws + OFF_R1);
  fill_zero16_kernel<<<nb(M2 * K2P / 16, T), T, 0, stream>>>((uint4*)a2, M2 * K2P / 16);
  im2col3x3_kernel<<<nb(M2 * 9, T), T, 0, stream>>>(hb2, a2, 1024, 14, 14, 48, K2P);
  float* c2 = (float*)(ws + OFF_R0);
  bgemm_iu8_kernel<<<dim3(128 / 128, (unsigned)(M2 / 64)), T, 0, stream>>>(a2, b2p, b2, c2, (int)M2, 128, K2P);
  fill_zero16_kernel<<<4, T, 0, stream>>>((uint4*)st, 1024);
  bn_stats_kernel<<<dim3(128, 64), T, 0, stream>>>(c2, st, M2, 128);
  bn_htanh_bin_kernel<<<nb(M2 * 128, T), T, 0, stream>>>(c2, st, g2, be2, hb3, M2, 128);

  signed char* a3 = (signed char*)(ws + OFF_R1);
  im2col3x3_kernel<<<nb(M2 * 9, T), T, 0, stream>>>(hb3, a3, 1024, 14, 14, 128, K3P);
  float* c3 = (float*)(ws + OFF_R0);
  bgemm_iu8_kernel<<<dim3(256 / 128, (unsigned)(M2 / 64)), T, 0, stream>>>(a3, b3p, b3, c3, (int)M2, 256, K3P);
  float* c3p = (float*)(ws + OFF_R1);
  maxpool2_nhwc_kernel<<<nb(50176LL * 256, T), T, 0, stream>>>(c3, c3p, 1024, 14, 14, 256);
  fill_zero16_kernel<<<4, T, 0, stream>>>((uint4*)st, 1024);
  bn_stats_kernel<<<dim3(256, 64), T, 0, stream>>>(c3p, st, 50176, 256);
  signed char* a4 = (signed char*)(ws + OFF_R0);
  bn_htanh_bin_reorder_kernel<<<nb(50176LL * 256, T), T, 0, stream>>>(c3p, st, g3, be3, a4);

  float* c4 = (float*)(ws + OFF_R1);
  bgemm_iu8_kernel<<<dim3(2048 / 128, 1024 / 64), T, 0, stream>>>(a4, b4p, bf1, c4, 1024, 2048, (int)K4);
  fill_zero16_kernel<<<4, T, 0, stream>>>((uint4*)st, 1024);
  bn_stats_kernel<<<dim3(2048, 64), T, 0, stream>>>(c4, st, 1024, 2048);
  bn_htanh_f_kernel<<<nb(1024LL * 2048, T), T, 0, stream>>>(c4, st, g4, be4, 1024, 2048);

  fc2_wmma_kernel<<<8, T, 0, stream>>>(c4, wf2, bf2, outp);
  logsoftmax10_kernel<<<4, T, 0, stream>>>(outp, 1024);
}
